// Cross_attention_2078764172179
// MI455X (gfx1250) — matrix operand order repaired and checked by reading, not compiled, not run on hardware
//
#include <hip/hip_runtime.h>
#include <hip/hip_bf16.h>
#include <math.h>

// ---------------------------------------------------------------------------
// Types
// ---------------------------------------------------------------------------
typedef _Float16 half_t;
typedef _Float16 v4h  __attribute__((ext_vector_type(4)));
typedef _Float16 v8h  __attribute__((ext_vector_type(8)));
typedef _Float16 v16h __attribute__((ext_vector_type(16)));
typedef float    v8f  __attribute__((ext_vector_type(8)));
typedef unsigned int u32;
typedef u32 u32x4 __attribute__((ext_vector_type(4)));
typedef int i32x4 __attribute__((ext_vector_type(4)));
typedef int i32x8 __attribute__((ext_vector_type(8)));

#define HW   4096
#define CC   128
#define NH   4
#define HD   32
#define NB   4
#define SEGH 131072   // halves per (b,h) segment (4096*32)

#if defined(__has_builtin)
#if __has_builtin(__builtin_amdgcn_tensor_load_to_lds)
#define HAVE_TDM 1
#endif
#endif
#ifndef HAVE_TDM
#define HAVE_TDM 0
#endif

// ---------------------------------------------------------------------------
// Fragment helpers (contiguous k order used by this file; A and B match)
// A 16x32 f16: lane L -> row = L&15; elems 0..15 -> k = 2*base..2*base+15,
//              base = (L<16)?0:8
// B 32x16 f16: lane L -> col = L&15; elems 0..15 -> k = kb..kb+15, kb=(L<16)?0:16
// C 16x16 f32: lane L -> col = L&15; vgpr i -> row = i + ((L<16)?0:8)
// ---------------------------------------------------------------------------
static __device__ inline v16h a_frag(const half_t* row, int base) {
  v8h lo = *(const v8h*)(row + 2 * base);
  v8h hi = *(const v8h*)(row + 2 * base + 8);
  return __builtin_shufflevector(lo, hi, 0,1,2,3,4,5,6,7,8,9,10,11,12,13,14,15);
}
static __device__ inline v16h b_frag(const half_t* p) {
  v8h lo = *(const v8h*)p;
  v8h hi = *(const v8h*)(p + 8);
  return __builtin_shufflevector(lo, hi, 0,1,2,3,4,5,6,7,8,9,10,11,12,13,14,15);
}
static __device__ inline v8f wmma16(v16h a, v16h b, v8f c) {
  return __builtin_amdgcn_wmma_f32_16x16x32_f16(false, a, false, b, (short)0, c,
                                                false, false);
}

// ---------------------------------------------------------------------------
// Tensor Data Mover: build D# (groups 0/1) per CDNA5 ISA
// 2-byte elements, optional LDS padding (pad_interval/pad_amount codes).
// This toolchain's builtin is the 6-arg form:
//   (u32x4 g0, i32x8 g1, i32x4 g2, i32x4 g3, i32x8 extra, i32 cpol)
// ---------------------------------------------------------------------------
#if HAVE_TDM
static __device__ inline void tdm_issue(void* ldsp, const void* g,
    u32 tile_d0, u32 tile_d1, u32 tensor_d0, u32 tensor_d1,
    u32 stride0, u32 padi, u32 pada, u32 pad_en)
{
  unsigned long long ga = (unsigned long long)(size_t)g;
  u32 ldso = (u32)(size_t)ldsp;
  u32x4 g0;
  g0[0] = 1u;                                            // count=1, no gather
  g0[1] = ldso;                                          // lds_addr
  g0[2] = (u32)ga;                                       // global_addr[31:0]
  g0[3] = ((u32)(ga >> 32) & 0x1FFFFFFu) | (2u << 30);   // addr[56:32] | type=2
  i32x8 g1;
  g1[0] = (int)((1u << 16) |                             // data_size = 2B
                (pad_en ? ((1u << 20) | (padi << 22) | (pada << 25)) : 0u));
  g1[1] = (int)((tensor_d0 & 0xFFFFu) << 16);
  g1[2] = (int)(((tensor_d0 >> 16) & 0xFFFFu) | ((tensor_d1 & 0xFFFFu) << 16));
  g1[3] = (int)(((tensor_d1 >> 16) & 0xFFFFu) | ((tile_d0 & 0xFFFFu) << 16));
  g1[4] = (int)(tile_d1 & 0xFFFFu);                      // tile_dim1 (tile_dim2=0)
  g1[5] = (int)stride0;                                  // tensor_dim0_stride lo
  g1[6] = 0;
  g1[7] = 0;
  i32x4 z4 = {0, 0, 0, 0};
  i32x8 z8 = {0, 0, 0, 0, 0, 0, 0, 0};
  __builtin_amdgcn_tensor_load_to_lds(g0, g1, z4, z4, z8, 0);
}
#endif

static __device__ inline void wait_tensorcnt_le2() {
#if defined(__has_builtin) && __has_builtin(__builtin_amdgcn_s_wait_tensorcnt)
  __builtin_amdgcn_s_wait_tensorcnt(2);
#else
  asm volatile("s_wait_tensorcnt 0x2" ::: "memory");
#endif
}

// ---------------------------------------------------------------------------
// Kernel: fp32 -> f16 cast (weights)
// ---------------------------------------------------------------------------
__global__ void cvt_kernel(const float* __restrict__ s, half_t* __restrict__ d, int n) {
  int i = blockIdx.x * 256 + threadIdx.x;
  if (i < n) d[i] = (half_t)s[i];
}

// ---------------------------------------------------------------------------
// Kernel: GroupNorm (16 groups of 8 channels).  One block per (b, group).
// Writes fp32 normalized (residual path) and f16 (GEMM path).
// ---------------------------------------------------------------------------
__global__ __launch_bounds__(256) void gn_kernel(
    const float* __restrict__ x, const float* __restrict__ w,
    const float* __restrict__ bb, float* __restrict__ y32,
    half_t* __restrict__ y16)
{
  const int blk = blockIdx.x;
  const int b = blk >> 4, g = blk & 15;
  const int tid = threadIdx.x;
  const size_t base = ((size_t)b * CC + g * 8) * HW;
  const float4* xv = (const float4*)(x + base);

  float s = 0.f, s2 = 0.f;
  for (int i = tid; i < 8192; i += 256) {
    float4 v = xv[i];
    s  += v.x + v.y + v.z + v.w;
    s2 += v.x * v.x + v.y * v.y + v.z * v.z + v.w * v.w;
  }
  __shared__ float red[16];
  #pragma unroll
  for (int off = 16; off > 0; off >>= 1) {
    s  += __shfl_down(s, off);
    s2 += __shfl_down(s2, off);
  }
  if ((tid & 31) == 0) { red[(tid >> 5) * 2] = s; red[(tid >> 5) * 2 + 1] = s2; }
  __syncthreads();
  if (tid == 0) {
    float ts = 0.f, t2 = 0.f;
    #pragma unroll
    for (int i = 0; i < 8; ++i) { ts += red[i * 2]; t2 += red[i * 2 + 1]; }
    float mean = ts * (1.f / 32768.f);
    float var  = t2 * (1.f / 32768.f) - mean * mean;
    red[0] = mean;
    red[1] = rsqrtf(var + 1e-5f);
  }
  __syncthreads();
  const float mean = red[0], inv = red[1];

  float4* y32v = (float4*)(y32 + base);
  v4h*    y16v = (v4h*)(y16 + base);
  const float* wp = w + g * 8;
  const float* bp = bb + g * 8;
  for (int i = tid; i < 8192; i += 256) {
    int c = i >> 10;
    float gw = wp[c] * inv;
    float gb = bp[c] - mean * gw;
    float4 v = xv[i];
    float4 o;
    o.x = v.x * gw + gb; o.y = v.y * gw + gb;
    o.z = v.z * gw + gb; o.w = v.w * gw + gb;
    y32v[i] = o;
    v4h hh; hh[0] = (half_t)o.x; hh[1] = (half_t)o.y;
    hh[2] = (half_t)o.z; hh[3] = (half_t)o.w;
    y16v[i] = hh;
  }
}

// ---------------------------------------------------------------------------
// Kernel: QKV projection GEMM  y[o,p] = sum_c W[o,c] * x[c,p]  (per batch)
// Block tile 64(M) x 128(N); 8 waves; x tile transposed into padded LDS.
// Epilogue scatters rows into Q^T(HWx32) / K^T(HWx32) / V(32xHW) per (b,head).
// ---------------------------------------------------------------------------
__global__ __launch_bounds__(256) void qkv_kernel(
    const half_t* __restrict__ Wf16,   // 384 x 128
    const half_t* __restrict__ X16,    // (B, 128, 4096)
    half_t* __restrict__ qkv)          // per (b,h): [Q^T | K^T | V]
{
  __shared__ half_t sXT[128 * 136];    // x^T tile, padded rows

  const int tid  = threadIdx.x;
  const int wave = tid >> 5, lane = tid & 31, nl = lane & 15;
  const int hl = lane >> 4, kb = hl * 16, ab = hl * 8;
  const int waveM = wave & 3, waveN = wave >> 2;
  const int b  = blockIdx.z;
  const int p0 = blockIdx.y * 128;
  const int mtile = blockIdx.x * 64 + waveM * 16;

  // cooperative transpose: x (128c x 128p window) -> sXT[p][c]
  const half_t* Xb = X16 + (size_t)b * (CC * HW);
  #pragma unroll
  for (int it = 0; it < 8; ++it) {
    int idx = it * 2048 + tid * 8;
    int c = idx >> 7;
    int p = idx & 127;
    v8h v = *(const v8h*)(Xb + (size_t)c * HW + p0 + p);
    #pragma unroll
    for (int j = 0; j < 8; ++j) sXT[(p + j) * 136 + c] = v[j];
  }
  __syncthreads();

  // A fragments straight from L2-resident f16 weights
  v16h af[4];
  const half_t* wrow = Wf16 + (size_t)(mtile + nl) * 128;
  #pragma unroll
  for (int kc = 0; kc < 4; ++kc) af[kc] = a_frag(wrow + kc * 32, ab);

  const int h   = mtile / 96;
  const int r96 = mtile % 96;
  const int seg = r96 / 32;          // 0=Q, 1=K, 2=V (uniform per 16-row tile)
  const int rb  = r96 % 32;
  half_t* segbase = qkv + (size_t)((b * NH + h) * 3 + seg) * SEGH;

  #pragma unroll
  for (int ns = 0; ns < 4; ++ns) {
    const int pl = waveN * 64 + ns * 16 + nl;
    v8f acc = {};
    #pragma unroll
    for (int kc = 0; kc < 4; ++kc) {
      v16h bf = b_frag(&sXT[pl * 136 + kc * 32 + kb]);
      acc = wmma16(af[kc], bf, acc);
    }
    const int p = p0 + pl;
    if (seg < 2) {                    // Q^T / K^T: contiguous v8h store
      v8h hv;
      #pragma unroll
      for (int i = 0; i < 8; ++i) hv[i] = (half_t)acc[i];
      *(v8h*)(segbase + (size_t)p * HD + rb + ab) = hv;
    } else {                          // V (d x HW): scattered stores
      #pragma unroll
      for (int i = 0; i < 8; ++i)
        segbase[(size_t)(rb + i + ab) * HW + p] = (half_t)acc[i];
    }
  }
}

// ---------------------------------------------------------------------------
// Kernel: flash cross-attention.  One block = 64 queries (4 waves x 16),
// streams 4096 keys in 64-key tiles.  K/V tiles staged in LDS by the TDM
// (double buffered, padded rows), P goes through an LDS round trip to get
// A-fragment layout.  Output written as O^T (HW x 128) per batch.
// ---------------------------------------------------------------------------
#define LDK 40
#define LDV 72
#define LDP 72

__global__ __launch_bounds__(128) void attn_kernel(
    const half_t* __restrict__ qkvQ,   // Q source (other stream)
    const half_t* __restrict__ qkvKV,  // K,V source
    half_t* __restrict__ OT)           // (B, HW, 128)
{
  __shared__ half_t sK[2][64 * LDK];   // K^T tile: 64 keys x 32d (padded)
  __shared__ half_t sV[2][32 * LDV];   // V tile:   32 d  x 64 keys (padded)
  __shared__ half_t sP[4][16 * LDP];   // per-wave P staging 16 x 64

  const int wave = threadIdx.x >> 5;
  const int lane = threadIdx.x & 31;
  const int nl = lane & 15;
  const int hl = lane >> 4;
  const int kb = hl * 16, ab = hl * 8;

  const int h = blockIdx.y, b = blockIdx.z;
  const int i0 = blockIdx.x * 64 + wave * 16;

  const size_t bh = (size_t)(b * NH + h) * 3 * SEGH;
  const half_t* Qb = qkvQ  + bh;              // Q^T (4096 x 32)
  const half_t* Kb = qkvKV + bh + SEGH;       // K^T (4096 x 32)
  const half_t* Vb = qkvKV + bh + 2 * SEGH;   // V   (32 x 4096)

  const v16h qa = a_frag(Qb + (size_t)(i0 + nl) * HD, ab);

  v8f oacc0 = {}, oacc1 = {};
  float m2[8], l[8];
  #pragma unroll
  for (int i = 0; i < 8; ++i) { m2[i] = -1e30f; l[i] = 0.f; }

  const float SC = 0.088388347648318447f * 1.4426950408889634f; // scale*log2(e)

#if HAVE_TDM
  if (wave == 0) {
    // K: 2D (32 x 64rows), row=16 DW -> padi=3; pad 4 DW -> pada=3 (stride 40)
    tdm_issue(&sK[0][0], Kb,            32, 64, 32, HW, 32, 3, 3, 1);
    // V: 2D (64 x 32rows), row=32 DW -> padi=4; pad 4 DW -> pada=3 (stride 72)
    tdm_issue(&sV[0][0], Vb,            64, 32, HW, 32, HW, 4, 3, 1);
    tdm_issue(&sK[1][0], Kb + 64 * HD,  32, 64, 32, HW, 32, 3, 3, 1);
    tdm_issue(&sV[1][0], Vb + 64,       64, 32, HW, 32, HW, 4, 3, 1);
  }
#endif

  for (int t = 0; t < 64; ++t) {
#if HAVE_TDM
    const int buf = t & 1;
    if (wave == 0) wait_tensorcnt_le2();
    __syncthreads();
#else
    const int buf = 0;
    __syncthreads();
    for (int idx = threadIdx.x * 8; idx < 64 * 32; idx += 128 * 8) {
      int r = idx >> 5, c = idx & 31;
      *(v8h*)&sK[0][r * LDK + c] = *(const v8h*)(Kb + (size_t)(t * 64 + r) * HD + c);
    }
    for (int idx = threadIdx.x * 8; idx < 32 * 64; idx += 128 * 8) {
      int r = idx >> 6, c = idx & 63;
      *(v8h*)&sV[0][r * LDV + c] = *(const v8h*)(Vb + (size_t)r * HW + t * 64 + c);
    }
    __syncthreads();
#endif

    // S = Q K^T for the 4 key sub-tiles
    v8f sv[4];
    #pragma unroll
    for (int s = 0; s < 4; ++s) {
      v16h kf = b_frag(&sK[buf][(s * 16 + nl) * LDK + kb]);
      v8f z = {};
      sv[s] = wmma16(qa, kf, z);
    }

    // online softmax (base-2 domain); row stats replicated via shfl_xor
    float mx[8];
    #pragma unroll
    for (int i = 0; i < 8; ++i) {
      float a0 = sv[0][i] * SC, a1 = sv[1][i] * SC;
      float a2 = sv[2][i] * SC, a3 = sv[3][i] * SC;
      sv[0][i] = a0; sv[1][i] = a1; sv[2][i] = a2; sv[3][i] = a3;
      mx[i] = fmaxf(fmaxf(a0, a1), fmaxf(a2, a3));
    }
    #pragma unroll
    for (int off = 1; off < 16; off <<= 1) {
      #pragma unroll
      for (int i = 0; i < 8; ++i) mx[i] = fmaxf(mx[i], __shfl_xor(mx[i], off));
    }
    float corr[8], ls[8];
    #pragma unroll
    for (int i = 0; i < 8; ++i) {
      float mn = fmaxf(m2[i], mx[i]);
      corr[i] = exp2f(m2[i] - mn);
      m2[i] = mn;
      float p0 = exp2f(sv[0][i] - mn), p1 = exp2f(sv[1][i] - mn);
      float p2 = exp2f(sv[2][i] - mn), p3 = exp2f(sv[3][i] - mn);
      sv[0][i] = p0; sv[1][i] = p1; sv[2][i] = p2; sv[3][i] = p3;
      ls[i] = (p0 + p1) + (p2 + p3);
    }
    #pragma unroll
    for (int off = 1; off < 16; off <<= 1) {
      #pragma unroll
      for (int i = 0; i < 8; ++i) ls[i] += __shfl_xor(ls[i], off);
    }
    #pragma unroll
    for (int i = 0; i < 8; ++i) {
      l[i] = l[i] * corr[i] + ls[i];
      oacc0[i] *= corr[i];
      oacc1[i] *= corr[i];
    }

    // P: C-layout -> LDS -> A-layout
    half_t* sp = sP[wave];
    #pragma unroll
    for (int s = 0; s < 4; ++s) {
      #pragma unroll
      for (int i = 0; i < 8; ++i)
        sp[(i + ab) * LDP + s * 16 + nl] = (half_t)sv[s][i];
    }
    v16h pf0 = a_frag(&sp[nl * LDP +  0], ab);
    v16h pf1 = a_frag(&sp[nl * LDP + 32], ab);

    // O^T += P * V^T  (two 16-wide d-column tiles, two 32-key chunks)
    {
      v16h vf00 = b_frag(&sV[buf][(0 * 16 + nl) * LDV +  0 + kb]);
      v16h vf01 = b_frag(&sV[buf][(0 * 16 + nl) * LDV + 32 + kb]);
      oacc0 = wmma16(pf0, vf00, oacc0);
      oacc0 = wmma16(pf1, vf01, oacc0);
      v16h vf10 = b_frag(&sV[buf][(1 * 16 + nl) * LDV +  0 + kb]);
      v16h vf11 = b_frag(&sV[buf][(1 * 16 + nl) * LDV + 32 + kb]);
      oacc1 = wmma16(pf0, vf10, oacc1);
      oacc1 = wmma16(pf1, vf11, oacc1);
    }

    __syncthreads();
#if HAVE_TDM
    if (wave == 0 && t + 2 < 64) {
      const int tt = t + 2;
      tdm_issue(&sK[buf][0], Kb + (size_t)tt * 64 * HD, 32, 64, 32, HW, 32, 3, 3, 1);
      tdm_issue(&sV[buf][0], Vb + tt * 64,              64, 32, HW, 32, HW, 4, 3, 1);
    }
#endif
  }

  // normalize and store O^T
  half_t* outp = OT + (size_t)b * ((size_t)HW * CC);
  #pragma unroll
  for (int i = 0; i < 8; ++i) {
    float inv = 1.0f / l[i];
    int qi = i0 + i + ab;
    outp[(size_t)qi * CC + h * HD + nl]      = (half_t)(oacc0[i] * inv);
    outp[(size_t)qi * CC + h * HD + 16 + nl] = (half_t)(oacc1[i] * inv);
  }
}

// ---------------------------------------------------------------------------
// Kernel: output projection + bias + residual(normalized input), fp32 out.
// B-operand = O^T (HW x 128) read directly from global (fragment-friendly).
// ---------------------------------------------------------------------------
__global__ __launch_bounds__(256) void proj_kernel(
    const half_t* __restrict__ Wo16,   // 128 x 128 f16
    const half_t* __restrict__ OTb,    // (B, HW, 128)
    const float* __restrict__ bias,    // 128
    const float* __restrict__ resid,   // (B, 128, HW) normalized input
    float* __restrict__ out)           // (B, 128, HW)
{
  const int tid  = threadIdx.x;
  const int wave = tid >> 5, lane = tid & 31, nl = lane & 15;
  const int hl = lane >> 4, kb = hl * 16, ab = hl * 8;
  const int waveM = wave & 3, waveN = wave >> 2;
  const int b  = blockIdx.z;
  const int p0 = blockIdx.y * 128;
  const int mtile = blockIdx.x * 64 + waveM * 16;

  const half_t* ot = OTb + (size_t)b * ((size_t)HW * CC);

  v16h af[4];
  const half_t* wrow = Wo16 + (size_t)(mtile + nl) * 128;
  #pragma unroll
  for (int kc = 0; kc < 4; ++kc) af[kc] = a_frag(wrow + kc * 32, ab);

  #pragma unroll
  for (int ns = 0; ns < 4; ++ns) {
    const int p = p0 + waveN * 64 + ns * 16 + nl;
    v8f acc = {};
    #pragma unroll
    for (int kc = 0; kc < 4; ++kc) {
      v16h bf = b_frag(ot + (size_t)p * CC + kc * 32 + kb);
      acc = wmma16(af[kc], bf, acc);
    }
    #pragma unroll
    for (int i = 0; i < 8; ++i) {
      const int o = mtile + i + ab;
      const size_t idx = ((size_t)b * CC + o) * HW + p;
      out[idx] = acc[i] + bias[o] + resid[idx];
    }
  }
}

// ---------------------------------------------------------------------------
// Host launcher
// ---------------------------------------------------------------------------
extern "C" void kernel_launch(void* const* d_in, const int* in_sizes, int n_in,
                              void* d_out, int out_size, void* d_ws, size_t ws_size,
                              hipStream_t stream) {
  (void)in_sizes; (void)n_in; (void)out_size; (void)ws_size;

  const float* x_A     = (const float*)d_in[0];
  const float* x_B     = (const float*)d_in[1];
  const float* gnA_w   = (const float*)d_in[2];
  const float* gnA_b   = (const float*)d_in[3];
  const float* gnB_w   = (const float*)d_in[4];
  const float* gnB_b   = (const float*)d_in[5];
  const float* W_qkv_A = (const float*)d_in[6];
  const float* W_qkv_B = (const float*)d_in[7];
  const float* W_out_A = (const float*)d_in[8];
  const float* b_out_A = (const float*)d_in[9];
  const float* W_out_B = (const float*)d_in[10];
  const float* b_out_B = (const float*)d_in[11];

  char* ws = (char*)d_ws;
  float*  xnA32 = (float*)(ws + 0);                 //  8 MB
  float*  xnB32 = (float*)(ws + 8388608);           //  8 MB
  half_t* xnA16 = (half_t*)(ws + 16777216);         //  4 MB
  half_t* xnB16 = (half_t*)(ws + 20971520);         //  4 MB
  half_t* qkvA  = (half_t*)(ws + 25165824);         // 12 MB
  half_t* qkvB  = (half_t*)(ws + 37748736);         // 12 MB
  half_t* oA    = (half_t*)(ws + 50331648);         //  4 MB (O^T)
  half_t* oB    = (half_t*)(ws + 54525952);         //  4 MB (O^T)
  half_t* WqA16 = (half_t*)(ws + 58720256);
  half_t* WqB16 = WqA16 + 49152;
  half_t* WoA16 = WqB16 + 49152;
  half_t* WoB16 = WoA16 + 16384;

  // weights -> f16
  cvt_kernel<<<192, 256, 0, stream>>>(W_qkv_A, WqA16, 49152);
  cvt_kernel<<<192, 256, 0, stream>>>(W_qkv_B, WqB16, 49152);
  cvt_kernel<<< 64, 256, 0, stream>>>(W_out_A, WoA16, 16384);
  cvt_kernel<<< 64, 256, 0, stream>>>(W_out_B, WoB16, 16384);

  // group norm
  gn_kernel<<<64, 256, 0, stream>>>(x_A, gnA_w, gnA_b, xnA32, xnA16);
  gn_kernel<<<64, 256, 0, stream>>>(x_B, gnB_w, gnB_b, xnB32, xnB16);

  // qkv projections
  qkv_kernel<<<dim3(6, 32, 4), 256, 0, stream>>>(WqA16, xnA16, qkvA);
  qkv_kernel<<<dim3(6, 32, 4), 256, 0, stream>>>(WqB16, xnB16, qkvB);

  // cross attention: out_A = attn(qB, kA, vA), out_B = attn(qA, kB, vB)
  attn_kernel<<<dim3(64, 4, 4), 128, 0, stream>>>(qkvB, qkvA, oA);
  attn_kernel<<<dim3(64, 4, 4), 128, 0, stream>>>(qkvA, qkvB, oB);

  // output projection + bias + normalized residual
  proj_kernel<<<dim3(2, 32, 4), 256, 0, stream>>>(WoA16, oA, b_out_A, xnA32,
                                                  (float*)d_out);
  proj_kernel<<<dim3(2, 32, 4), 256, 0, stream>>>(WoB16, oB, b_out_B, xnB32,
                                                  (float*)d_out + 2097152);
}
